// MaskedMultiHeadedAttn_42941083025564
// MI455X (gfx1250) — matrix operand order repaired and checked by reading, not compiled, not run on hardware
//
#include <hip/hip_runtime.h>
#include <hip/hip_bf16.h>

#define NH    16
#define HS    64
#define HID   1024
#define SEQ   2048
#define BATCH 4

typedef __attribute__((ext_vector_type(16))) __bf16 v16bf;
typedef __attribute__((ext_vector_type(8)))  __bf16 v8bf;
typedef __attribute__((ext_vector_type(8)))  float  v8f;
typedef __attribute__((ext_vector_type(4)))  unsigned int u32x4;
typedef __attribute__((ext_vector_type(8)))  int i32x8;
typedef __attribute__((ext_vector_type(4)))  int i32x4;

#if __has_builtin(__builtin_amdgcn_tensor_load_to_lds)
#define USE_TDM 1
#else
#define USE_TDM 0
#endif

__device__ __forceinline__ v8f wmma_bf16(v16bf a, v16bf b, v8f c) {
  // D = A(16x32 bf16) * B(32x16 bf16) + C(16x16 f32)
  return __builtin_amdgcn_wmma_f32_16x16x32_bf16(false, a, false, b, (short)0, c,
                                                 false, false);
}

__device__ __forceinline__ v16bf combine(v8bf lo, v8bf hi) {
  return __builtin_shufflevector(lo, hi, 0, 1, 2, 3, 4, 5, 6, 7,
                                 8, 9, 10, 11, 12, 13, 14, 15);
}

// A-fragment (16x32, MxK): lane holds row M=lane%16; contiguous K chunk
// {koff..koff+15}, koff=(lane/16)*16.  Two 16B loads.
__device__ __forceinline__ v16bf load_a_frag(const __bf16* row, int koff) {
  v8bf lo = *(const v8bf*)(row + koff);
  v8bf hi = *(const v8bf*)(row + koff + 8);
  return combine(lo, hi);
}
// B-fragment (32x16, KxN): lane holds col N=lane%16; contiguous K chunk
// {koff..koff+15}, koff=(lane/16)*16.  Two 16B loads.
__device__ __forceinline__ v16bf load_b_frag(const __bf16* row, int koff) {
  v8bf lo = *(const v8bf*)(row + koff);
  v8bf hi = *(const v8bf*)(row + koff + 8);
  return combine(lo, hi);
}

#if USE_TDM
// Issue a 2D TDM tile load (bf16 elements) global -> LDS.  D# packing per
// CDNA5 ISA §8.3-8.5: group0 = {count|flags, lds_addr, global_addr, type},
// group1 = {mask/data_size, tensor_dim0/1, tile_dim0/1/2, dim0_stride}.
// This toolchain exposes the 6-arg builtin (4 descriptor groups + spare + cpol).
__device__ __forceinline__ void tdm_load_2d_bf16(
    unsigned lds_addr, const void* gptr,
    unsigned tdim0, unsigned tdim1, unsigned stride0,
    unsigned tile0, unsigned tile1) {
  const unsigned long long ga = (unsigned long long)gptr;
  u32x4 g0;
  g0[0] = 1u;                                    // count=1, user descriptor
  g0[1] = lds_addr;                              // LDS byte address
  g0[2] = (unsigned)ga;                          // global_addr[31:0]
  g0[3] = (unsigned)((ga >> 32) & 0x01FFFFFFu)   // global_addr[56:32]
          | 0x80000000u;                         // type = 2 ("image")
  i32x8 g1;
  g1[0] = 0x00010000;                            // data_size=1 -> 2 bytes
  g1[1] = (int)(tdim0 << 16);                    // tensor_dim0[15:0]
  g1[2] = (int)((tdim0 >> 16) | (tdim1 << 16));  // dim0[31:16] | dim1[15:0]
  g1[3] = (int)((tdim1 >> 16) | (tile0 << 16));  // dim1[31:16] | tile_dim0
  g1[4] = (int)(tile1 & 0xFFFFu);                // tile_dim1 (tile_dim2=0)
  g1[5] = (int)stride0;                          // tensor_dim0_stride[31:0]
  g1[6] = 0;
  g1[7] = 0;
  const i32x4 z4 = {0, 0, 0, 0};
  const i32x8 z8 = {0, 0, 0, 0, 0, 0, 0, 0};
  __builtin_amdgcn_tensor_load_to_lds(g0, g1, z4, z4, z8, 0);
}
#endif

// ---------------------------------------------------------------------------
// Kernel 0: one-time f32 -> bf16 conversion (x and weights), 8 elems/thread.
// ---------------------------------------------------------------------------
__global__ void __launch_bounds__(256) cvt_f32_bf16_kernel(
    const float* __restrict__ src, __bf16* __restrict__ dst, int n8) {
  const int i = (blockIdx.x * 256 + threadIdx.x);
  if (i < n8) {
    const float4 a = *(const float4*)(src + (size_t)i * 8);
    const float4 b = *(const float4*)(src + (size_t)i * 8 + 4);
    v8bf o;
    o[0] = (__bf16)a.x; o[1] = (__bf16)a.y; o[2] = (__bf16)a.z; o[3] = (__bf16)a.w;
    o[4] = (__bf16)b.x; o[5] = (__bf16)b.y; o[6] = (__bf16)b.z; o[7] = (__bf16)b.w;
    *(v8bf*)(dst + (size_t)i * 8) = o;
  }
}

// ---------------------------------------------------------------------------
// Kernel 1: fused QKV projection (all-bf16 operands). Each wave: 32x64 tile.
// ---------------------------------------------------------------------------
__global__ void __launch_bounds__(256) qkv_proj_kernel(
    const __bf16* __restrict__ xb,
    const __bf16* __restrict__ Wqb, const float* __restrict__ bq,
    const __bf16* __restrict__ Wkb, const float* __restrict__ bk,
    const __bf16* __restrict__ Wvb, const float* __restrict__ bv,
    __bf16* __restrict__ Qh, __bf16* __restrict__ Kh, __bf16* __restrict__ Vt) {
  const int lane = threadIdx.x & 31;
  const int wave = threadIdx.x >> 5;
  const int unit = blockIdx.x * 8 + wave;

  const int MT = (BATCH * SEQ) / 32;  // 256
  const int NT = HID / 64;            // 16
  const int which = unit / (MT * NT); // 0=Q 1=K 2=V
  const int rem   = unit % (MT * NT);
  const int mt    = rem % MT;
  const int nt    = rem / MT;

  const __bf16* W    = (which == 0) ? Wqb : (which == 1) ? Wkb : Wvb;
  const float*  bias = (which == 0) ? bq  : (which == 1) ? bk  : bv;

  const int l16   = lane & 15;
  const int hlf   = lane >> 4;
  const int akoff = hlf * 16;
  const int bkoff = hlf * 16;
  const int nbase = nt * 64;

  const __bf16* arow0 = xb + (size_t)(mt * 32 + l16) * HID;
  const __bf16* arow1 = xb + (size_t)(mt * 32 + 16 + l16) * HID;
  const __bf16* wrow[4];
#pragma unroll
  for (int s = 0; s < 4; ++s)
    wrow[s] = W + (size_t)(nbase + s * 16 + l16) * HID;

  v8f acc[2][4] = {};
#pragma unroll 2
  for (int kk = 0; kk < HID; kk += 32) {
    v16bf a0 = load_a_frag(arow0 + kk, akoff);
    v16bf a1 = load_a_frag(arow1 + kk, akoff);
    v16bf bm[4];
#pragma unroll
    for (int s = 0; s < 4; ++s) bm[s] = load_b_frag(wrow[s] + kk, bkoff);
#pragma unroll
    for (int s = 0; s < 4; ++s) {
      acc[0][s] = wmma_bf16(a0, bm[s], acc[0][s]);
      acc[1][s] = wmma_bf16(a1, bm[s], acc[1][s]);
    }
  }

#pragma unroll
  for (int i = 0; i < 2; ++i) {
#pragma unroll
    for (int r = 0; r < 8; ++r) {
      const int m = i * 16 + r + 8 * hlf;
      const int t = mt * 32 + m;
      const int bb = t / SEQ, sidx = t % SEQ;
#pragma unroll
      for (int s = 0; s < 4; ++s) {
        const int o = nbase + s * 16 + l16;
        const float v = acc[i][s][r] + bias[o];
        const int head = o >> 6, d = o & 63;
        if (which == 2) {
          Vt[(((size_t)bb * NH + head) * HS + d) * SEQ + sidx] = (__bf16)v;
        } else {
          __bf16* dst = (which == 0) ? Qh : Kh;
          dst[(((size_t)bb * NH + head) * SEQ + sidx) * HS + d] = (__bf16)v;
        }
      }
    }
  }
}

// ---------------------------------------------------------------------------
// Kernel 2: causal flash attention.  128 threads = 4 waves per workgroup;
// the workgroup covers 64 queries (16 per wave) and shares double-buffered
// K/V key-block tiles in LDS, staged by the Tensor Data Mover (wave 0).
// ---------------------------------------------------------------------------
__global__ void __launch_bounds__(128) attn_kernel(
    const __bf16* __restrict__ Qh, const __bf16* __restrict__ Kh,
    const __bf16* __restrict__ Vt, __bf16* __restrict__ attnout) {
  __shared__ __align__(32) __bf16 Ktile[2][32 * 64];  // keys x d
  __shared__ __align__(32) __bf16 Vtile[2][64 * 32];  // d x keys (V^T)
  __shared__ __align__(32) __bf16 pbuf[4][16 * 32];   // per-wave P restage

  const int tid  = threadIdx.x;
  const int lane = tid & 31;
  const int wv   = tid >> 5;                  // 0..3
  const int qt64 = blockIdx.x & 31;           // SEQ/64 = 32 query blocks
  const int head = (blockIdx.x >> 5) & 15;
  const int bb   = blockIdx.x >> 9;
  const int qb64 = qt64 * 64;
  const int qbase = qb64 + wv * 16;

  const __bf16* Qbase = Qh + ((size_t)bb * NH + head) * SEQ * HS;
  const __bf16* Kbase = Kh + ((size_t)bb * NH + head) * SEQ * HS;
  const __bf16* Vbase = Vt + ((size_t)bb * NH + head) * HS * SEQ;

  const int l16   = lane & 15;
  const int hlf   = lane >> 4;
  const int akoff = hlf * 16;
  const int bkoff = hlf * 16;

  const __bf16* qrow = Qbase + (size_t)(qbase + l16) * HS;
  const v16bf qA0 = load_a_frag(qrow, akoff);
  const v16bf qA1 = load_a_frag(qrow + 32, akoff);

  v8f oacc[4] = {};
  float mrow[8], lrow[8];
#pragma unroll
  for (int r = 0; r < 8; ++r) { mrow[r] = -1e30f; lrow[r] = 0.0f; }

  // uniform causal bound for the whole workgroup (max query = qb64+63)
  const int nkb = qb64 / 32 + 2;

#if USE_TDM
  if (wv == 0) {  // prologue: TDM loads for block 0 into buffer 0
    tdm_load_2d_bf16((unsigned)(unsigned long long)(void*)&Ktile[0][0],
                     Kbase, HS, SEQ, HS, HS, 32);
    tdm_load_2d_bf16((unsigned)(unsigned long long)(void*)&Vtile[0][0],
                     Vbase, SEQ, HS, SEQ, 32, 64);
  }
#endif

  for (int kb = 0; kb < nkb; ++kb) {
    const int k0 = kb * 32;
#if USE_TDM
    const int cur = kb & 1;
    if (wv == 0) {
      if (kb + 1 < nkb) {  // prefetch next key block into the other buffer
        tdm_load_2d_bf16((unsigned)(unsigned long long)(void*)&Ktile[1 - cur][0],
                         Kbase + (size_t)(k0 + 32) * HS, HS, SEQ, HS, HS, 32);
        tdm_load_2d_bf16((unsigned)(unsigned long long)(void*)&Vtile[1 - cur][0],
                         Vbase + (k0 + 32), SEQ, HS, SEQ, 32, 64);
        __builtin_amdgcn_s_wait_tensorcnt(2);  // current block's 2 xfers done
      } else {
        __builtin_amdgcn_s_wait_tensorcnt(0);
      }
    }
    __syncthreads();
    const __bf16* Kt  = &Ktile[cur][0];
    const __bf16* Vtl = &Vtile[cur][0];
#else
    {  // cooperative staging fallback (128 threads copy 4KB + 4KB)
      const int kr = tid >> 2, kc = (tid & 3) * 16;
      *(v8bf*)&Ktile[0][kr * 64 + kc] =
          *(const v8bf*)(Kbase + (size_t)(k0 + kr) * HS + kc);
      *(v8bf*)&Ktile[0][kr * 64 + kc + 8] =
          *(const v8bf*)(Kbase + (size_t)(k0 + kr) * HS + kc + 8);
      const int vr = tid >> 1, vc = (tid & 1) * 16;
      *(v8bf*)&Vtile[0][vr * 32 + vc] =
          *(const v8bf*)(Vbase + (size_t)vr * SEQ + k0 + vc);
      *(v8bf*)&Vtile[0][vr * 32 + vc + 8] =
          *(const v8bf*)(Vbase + (size_t)vr * SEQ + k0 + vc + 8);
    }
    __syncthreads();
    const __bf16* Kt  = &Ktile[0][0];
    const __bf16* Vtl = &Vtile[0][0];
#endif

    // --- fragments from LDS ---
    v16bf kB[2][2], vB[4];
#pragma unroll
    for (int t2 = 0; t2 < 2; ++t2) {
      kB[t2][0] = load_b_frag(Kt + (t2 * 16 + l16) * 64, bkoff);
      kB[t2][1] = load_b_frag(Kt + (t2 * 16 + l16) * 64 + 32, bkoff);
    }
#pragma unroll
    for (int s = 0; s < 4; ++s)
      vB[s] = load_b_frag(Vtl + (s * 16 + l16) * 32, bkoff);

    // --- scores ---
    v8f sc[2];
#pragma unroll
    for (int t2 = 0; t2 < 2; ++t2) {
      v8f t = {};
      t = wmma_bf16(qA0, kB[t2][0], t);
      t = wmma_bf16(qA1, kB[t2][1], t);
      sc[t2] = t;
    }

    // --- online softmax ---
    float p0[8], p1[8], fr[8];
#pragma unroll
    for (int r = 0; r < 8; ++r) {
      const int q = qbase + r + 8 * hlf;
      float x0 = sc[0][r] * 0.125f;  // 1/sqrt(64)
      float x1 = sc[1][r] * 0.125f;
      if (q < k0 + l16)      x0 = -1e9f;
      if (q < k0 + 16 + l16) x1 = -1e9f;
      float rm = fmaxf(x0, x1);
      rm = fmaxf(rm, __shfl_xor(rm, 1, 32));
      rm = fmaxf(rm, __shfl_xor(rm, 2, 32));
      rm = fmaxf(rm, __shfl_xor(rm, 4, 32));
      rm = fmaxf(rm, __shfl_xor(rm, 8, 32));
      const float mnew = fmaxf(mrow[r], rm);
      const float f  = __expf(mrow[r] - mnew);
      const float e0 = __expf(x0 - mnew);
      const float e1 = __expf(x1 - mnew);
      float rs = e0 + e1;
      rs += __shfl_xor(rs, 1, 32);
      rs += __shfl_xor(rs, 2, 32);
      rs += __shfl_xor(rs, 4, 32);
      rs += __shfl_xor(rs, 8, 32);
      lrow[r] = lrow[r] * f + rs;
      mrow[r] = mnew;
      fr[r] = f; p0[r] = e0; p1[r] = e1;
    }
#pragma unroll
    for (int s = 0; s < 4; ++s)
#pragma unroll
      for (int r = 0; r < 8; ++r) oacc[s][r] *= fr[r];

    // --- restage P (same-wave LDS ops are in-order; no barrier needed) ---
    __bf16* pw = &pbuf[wv][0];
#pragma unroll
    for (int r = 0; r < 8; ++r) {
      const int m = r + 8 * hlf;
      pw[m * 32 + l16]      = (__bf16)p0[r];
      pw[m * 32 + 16 + l16] = (__bf16)p1[r];
    }
    const v16bf pA = load_a_frag(pw + l16 * 32, akoff);

    // --- O += P(16x32) * V(32x64) ---
#pragma unroll
    for (int s = 0; s < 4; ++s) oacc[s] = wmma_bf16(pA, vB[s], oacc[s]);

    __syncthreads();  // tiles fully consumed before next staging overwrites
  }

  // epilogue: normalize and store as bf16 in (B,S,H) for the output GEMM
#pragma unroll
  for (int r = 0; r < 8; ++r) {
    const int m = r + 8 * hlf;
    const int t = bb * SEQ + qbase + m;
    const float inv = 1.0f / lrow[r];
#pragma unroll
    for (int s = 0; s < 4; ++s) {
      const int col = head * HS + s * 16 + l16;
      attnout[(size_t)t * HID + col] = (__bf16)(oacc[s][r] * inv);
    }
  }
}

// ---------------------------------------------------------------------------
// Kernel 3: output projection. out = attn @ Wo^T + bo (f32 output).
// ---------------------------------------------------------------------------
__global__ void __launch_bounds__(256) out_proj_kernel(
    const __bf16* __restrict__ attnout,
    const __bf16* __restrict__ Wob, const float* __restrict__ bo,
    float* __restrict__ out) {
  const int lane = threadIdx.x & 31;
  const int wave = threadIdx.x >> 5;
  const int unit = blockIdx.x * 8 + wave;

  const int MT = (BATCH * SEQ) / 32;  // 256
  const int mt = unit % MT;
  const int nt = unit / MT;

  const int l16   = lane & 15;
  const int hlf   = lane >> 4;
  const int akoff = hlf * 16;
  const int bkoff = hlf * 16;
  const int nbase = nt * 64;

  const __bf16* arow0 = attnout + (size_t)(mt * 32 + l16) * HID;
  const __bf16* arow1 = attnout + (size_t)(mt * 32 + 16 + l16) * HID;
  const __bf16* wrow[4];
#pragma unroll
  for (int s = 0; s < 4; ++s)
    wrow[s] = Wob + (size_t)(nbase + s * 16 + l16) * HID;

  v8f acc[2][4] = {};
#pragma unroll 2
  for (int kk = 0; kk < HID; kk += 32) {
    v16bf a0 = load_a_frag(arow0 + kk, akoff);
    v16bf a1 = load_a_frag(arow1 + kk, akoff);
    v16bf bm[4];
#pragma unroll
    for (int s = 0; s < 4; ++s) bm[s] = load_b_frag(wrow[s] + kk, bkoff);
#pragma unroll
    for (int s = 0; s < 4; ++s) {
      acc[0][s] = wmma_bf16(a0, bm[s], acc[0][s]);
      acc[1][s] = wmma_bf16(a1, bm[s], acc[1][s]);
    }
  }

#pragma unroll
  for (int i = 0; i < 2; ++i) {
#pragma unroll
    for (int r = 0; r < 8; ++r) {
      const int t = mt * 32 + i * 16 + r + 8 * hlf;
#pragma unroll
      for (int s = 0; s < 4; ++s) {
        const int o = nbase + s * 16 + l16;
        out[(size_t)t * HID + o] = acc[i][s][r] + bo[o];
      }
    }
  }
}

extern "C" void kernel_launch(void* const* d_in, const int* in_sizes, int n_in,
                              void* d_out, int out_size, void* d_ws, size_t ws_size,
                              hipStream_t stream) {
  const float* x  = (const float*)d_in[0];
  const float* Wq = (const float*)d_in[1];
  const float* bq = (const float*)d_in[2];
  const float* Wk = (const float*)d_in[3];
  const float* bk = (const float*)d_in[4];
  const float* Wv = (const float*)d_in[5];
  const float* bv = (const float*)d_in[6];
  const float* Wo = (const float*)d_in[7];
  const float* bo = (const float*)d_in[8];
  float* out = (float*)d_out;

  char* ws = (char*)d_ws;
  const size_t matB = (size_t)BATCH * NH * SEQ * HS * sizeof(__bf16);  // 16 MB
  const size_t wB   = (size_t)HID * HID * sizeof(__bf16);              // 2 MB
  __bf16* Qh  = (__bf16*)(ws + 0 * matB);
  __bf16* Kh  = (__bf16*)(ws + 1 * matB);
  __bf16* Vt  = (__bf16*)(ws + 2 * matB);
  __bf16* Ah  = (__bf16*)(ws + 3 * matB);
  __bf16* xb  = (__bf16*)(ws + 4 * matB);
  __bf16* Wqb = (__bf16*)(ws + 5 * matB + 0 * wB);
  __bf16* Wkb = (__bf16*)(ws + 5 * matB + 1 * wB);
  __bf16* Wvb = (__bf16*)(ws + 5 * matB + 2 * wB);
  __bf16* Wob = (__bf16*)(ws + 5 * matB + 3 * wB);

  const int nx8 = BATCH * SEQ * HID / 8;
  const int nw8 = HID * HID / 8;
  cvt_f32_bf16_kernel<<<(nx8 + 255) / 256, 256, 0, stream>>>(x, xb, nx8);
  cvt_f32_bf16_kernel<<<(nw8 + 255) / 256, 256, 0, stream>>>(Wq, Wqb, nw8);
  cvt_f32_bf16_kernel<<<(nw8 + 255) / 256, 256, 0, stream>>>(Wk, Wkb, nw8);
  cvt_f32_bf16_kernel<<<(nw8 + 255) / 256, 256, 0, stream>>>(Wv, Wvb, nw8);
  cvt_f32_bf16_kernel<<<(nw8 + 255) / 256, 256, 0, stream>>>(Wo, Wob, nw8);

  qkv_proj_kernel<<<3 * 256 * 16 / 8, 256, 0, stream>>>(
      xb, Wqb, bq, Wkb, bk, Wvb, bv, Qh, Kh, Vt);

  // 4 waves / 64 queries per workgroup: B*NH*(SEQ/64) = 2048 blocks
  attn_kernel<<<BATCH * NH * (SEQ / 64), 128, 0, stream>>>(Qh, Kh, Vt, Ah);

  out_proj_kernel<<<256 * 16 / 8, 256, 0, stream>>>(Ah, Wob, bo, out);
}
